// TradeWos_2465311228064
// MI455X (gfx1250) — compile-verified, matrix operand order checked by reading, not run on hardware
//
#include <hip/hip_runtime.h>
#include <hip/hip_bf16.h>
#include <math.h>

// Problem dims (match reference)
#define Sdim 256
#define Bdim 8
#define Hdim 400
#define Vdim 16000
#define NSdim 30
#define Tdim 10
#define Gdim 3
#define NB   (NSdim * Bdim)     // 240
#define H3   (3 * Hdim)         // 1200
#define Kp   416                // H padded to multiple of 32 for WMMA K
#define SB   (Sdim * Bdim)      // 2048

typedef __attribute__((ext_vector_type(16))) _Float16 v16h;
typedef __attribute__((ext_vector_type(8)))  _Float16 v8h;
typedef __attribute__((ext_vector_type(8)))  float    v8f;

__device__ __forceinline__ float sigm(float x) { return 1.0f / (1.0f + __expf(-x)); }

__device__ __forceinline__ v16h load_frag(const _Float16* p)
{
    v8h lo = *(const v8h*)(p);
    v8h hi = *(const v8h*)(p + 16);
    return __builtin_shufflevector(lo, hi, 0,1,2,3,4,5,6,7,8,9,10,11,12,13,14,15);
}

// ---------------------------------------------------------------------------
// WMMA GEMM with B-fragment register reuse across MT M-tiles per wave:
//   C[M,N] = A[M,Kp] * Bw[N,Kp]^T (+ bias[N])
// A, Bw row-major f16, row stride Kp, zero-padded cols >= 400.
// Each wave computes an (MT*16) x 16 strip: per k-step one B fragment feeds
// MT v_wmma_f32_16x16x32_f16 ops -> cuts B (e.g. emb) L2 traffic by MT.
// ---------------------------------------------------------------------------
template <int MT>
__global__ __launch_bounds__(32) void wmma_gemm_mt(
    const _Float16* __restrict__ A,
    const _Float16* __restrict__ Bw,
    const float*    __restrict__ bias,
    float*          __restrict__ C,
    int M, int N, int Kpad)
{
    const int n0   = blockIdx.x * 16;
    const int m0   = blockIdx.y * (16 * MT);
    const int lane = threadIdx.x;
    const int half = lane >> 4;      // 0: K 0..7 & 16..23, 1: K 8..15 & 24..31
    const int l    = lane & 15;

    const _Float16* abase = A  + (size_t)(m0 + l) * Kpad + half * 8;
    const _Float16* brow  = Bw + (size_t)(n0 + l) * Kpad + half * 8;

    v8f acc[MT];
    #pragma unroll
    for (int mt = 0; mt < MT; ++mt) acc[mt] = (v8f){};

    for (int k = 0; k < Kpad; k += 32) {
        v16h bv = load_frag(brow + k);
        #pragma unroll
        for (int mt = 0; mt < MT; ++mt) {
            v16h av = load_frag(abase + (size_t)mt * 16 * Kpad + k);
            acc[mt] = __builtin_amdgcn_wmma_f32_16x16x32_f16(
                          false, av, false, bv, (short)0, acc[mt], false, false);
        }
    }
    const float bn = bias ? bias[n0 + l] : 0.0f;
    #pragma unroll
    for (int mt = 0; mt < MT; ++mt) {
        const int mbase = m0 + mt * 16 + half * 8;
        #pragma unroll
        for (int i = 0; i < 8; ++i)
            C[(size_t)(mbase + i) * N + (n0 + l)] = acc[mt][i] + bn;
    }
}

// ---------------------------------------------------------------------------
// Conversions / staging
// ---------------------------------------------------------------------------
// fp32 [N,400] -> f16 [N,416] zero padded
__global__ void cvt_pad_kernel(const float* __restrict__ src, _Float16* __restrict__ dst, int N)
{
    int idx = blockIdx.x * blockDim.x + threadIdx.x;
    if (idx >= N * Kp) return;
    int r = idx / Kp, c = idx % Kp;
    dst[idx] = (c < Hdim) ? (_Float16)src[r * Hdim + c] : (_Float16)0.0f;
}

// x_h[s*B+b, :] = f16(emb[input_seqs[s,b], :]), zero padded
__global__ void embed_x_kernel(const int* __restrict__ ids, const float* __restrict__ emb,
                               _Float16* __restrict__ xh)
{
    int idx = blockIdx.x * blockDim.x + threadIdx.x;
    if (idx >= SB * Kp) return;
    int r = idx / Kp, c = idx % Kp;
    xh[idx] = (c < Hdim) ? (_Float16)emb[(size_t)ids[r] * Hdim + c] : (_Float16)0.0f;
}

// ---------------------------------------------------------------------------
// Persistent bi-GRU scan: block 0 = forward (writes enc_out = 2*out_f, hf),
// block 1 = backward (writes hb only). gi (= x@Wi^T + bi) precomputed by WMMA.
// h and gh live in LDS; each step: gh = h@Wh^T + bh, then gate update.
// ---------------------------------------------------------------------------
__global__ __launch_bounds__(1024) void enc_scan_kernel(
    const float* __restrict__ gi_f, const float* __restrict__ gi_b,
    const float* __restrict__ Wh_f, const float* __restrict__ bh_f,
    const float* __restrict__ Wh_b, const float* __restrict__ bh_b,
    float* __restrict__ enc_out, float* __restrict__ hf, float* __restrict__ hb)
{
    __shared__ float h_lds[Bdim * Hdim];     // 12.8 KB
    __shared__ float gh_lds[Bdim * H3];      // 38.4 KB
    const int dir = blockIdx.x;
    const float* gi = dir ? gi_b : gi_f;
    const float* Wh = dir ? Wh_b : Wh_f;
    const float* bh = dir ? bh_b : bh_f;
    const int tid = threadIdx.x;

    for (int i = tid; i < Bdim * Hdim; i += 1024) h_lds[i] = 0.0f;
    __syncthreads();

    for (int step = 0; step < Sdim; ++step) {
        const int s = dir ? (Sdim - 1 - step) : step;
        // gh = h @ Wh^T + bh
        for (int idx = tid; idx < Bdim * H3; idx += 1024) {
            int b = idx / H3, n = idx % H3;
            const float* hr = h_lds + b * Hdim;
            const float* wr = Wh + (size_t)n * Hdim;
            float acc = bh[n];
            for (int k = 0; k < Hdim; ++k) acc += hr[k] * wr[k];
            gh_lds[idx] = acc;
        }
        __syncthreads();
        // gates + state update
        for (int idx = tid; idx < Bdim * Hdim; idx += 1024) {
            int b = idx / Hdim, j = idx % Hdim;
            const float* gir = gi + (size_t)(s * Bdim + b) * H3;
            const float* ghr = gh_lds + b * H3;
            float r  = sigm(gir[j] + ghr[j]);
            float z  = sigm(gir[Hdim + j] + ghr[Hdim + j]);
            float nn = tanhf(gir[2 * Hdim + j] + r * ghr[2 * Hdim + j]);
            float hn = (1.0f - z) * nn + z * h_lds[idx];
            h_lds[idx] = hn;
            if (dir == 0)  // enc_out[b,s,:] = out_f + out_f
                enc_out[((size_t)b * Sdim + s) * Hdim + j] = 2.0f * hn;
        }
        __syncthreads();
    }
    float* hout = dir ? hb : hf;
    for (int i = tid; i < Bdim * Hdim; i += 1024) hout[i] = h_lds[i];
}

// ---------------------------------------------------------------------------
// Decoder initial state: hidden = (hf+hb) broadcast over slots,
// dec_in = slot_emb broadcast over batch; f16 staged copies (zero padded).
// ---------------------------------------------------------------------------
__global__ void dec_init_kernel(const float* __restrict__ hf, const float* __restrict__ hb,
                                const float* __restrict__ slot_table,
                                const int* __restrict__ dom, const int* __restrict__ slot,
                                float* __restrict__ hidden, _Float16* __restrict__ hidden_h,
                                float* __restrict__ dec_in, _Float16* __restrict__ dec_in_h)
{
    int idx = blockIdx.x * blockDim.x + threadIdx.x;
    if (idx >= NB * Kp) return;
    int nb = idx / Kp, c = idx % Kp;
    int n = nb / Bdim, b = nb % Bdim;
    if (c < Hdim) {
        float hv = hf[b * Hdim + c] + hb[b * Hdim + c];
        float sv = slot_table[(size_t)dom[n] * Hdim + c] + slot_table[(size_t)slot[n] * Hdim + c];
        hidden[nb * Hdim + c] = hv;  hidden_h[idx] = (_Float16)hv;
        dec_in[nb * Hdim + c] = sv;  dec_in_h[idx] = (_Float16)sv;
    } else {
        hidden_h[idx] = (_Float16)0.0f;
        dec_in_h[idx] = (_Float16)0.0f;
    }
}

// GRU gate merge: h2 = (1-z)*n + z*h, updates hidden in place + f16 staging
__global__ void dec_gate_kernel(const float* __restrict__ gi, const float* __restrict__ gh,
                                float* __restrict__ hidden, _Float16* __restrict__ hidden_h)
{
    int idx = blockIdx.x * blockDim.x + threadIdx.x;
    if (idx >= NB * Hdim) return;
    int nb = idx / Hdim, j = idx % Hdim;
    const float* gir = gi + (size_t)nb * H3;
    const float* ghr = gh + (size_t)nb * H3;
    float r  = sigm(gir[j] + ghr[j]);
    float z  = sigm(gir[Hdim + j] + ghr[Hdim + j]);
    float nn = tanhf(gir[2 * Hdim + j] + r * ghr[2 * Hdim + j]);
    float h2 = (1.0f - z) * nn + z * hidden[idx];
    hidden[idx] = h2;
    hidden_h[nb * Kp + j] = (_Float16)h2;
}

// ---------------------------------------------------------------------------
// Attention: one WG per (slot,batch) row. scores -> softmax -> prob, context.
// ---------------------------------------------------------------------------
__global__ __launch_bounds__(256) void attn_kernel(
    const float* __restrict__ hidden, const float* __restrict__ enc_out,
    float* __restrict__ prob, float* __restrict__ context)
{
    __shared__ float hrow[Hdim];
    __shared__ float sc[Sdim];
    __shared__ float red[256];
    const int nb = blockIdx.x;
    const int b  = nb % Bdim;
    const int tid = threadIdx.x;

    for (int i = tid; i < Hdim; i += 256) hrow[i] = hidden[nb * Hdim + i];
    __syncthreads();

    const float* er = enc_out + ((size_t)b * Sdim + tid) * Hdim;
    float acc = 0.0f;
    for (int k = 0; k < Hdim; ++k) acc += hrow[k] * er[k];
    sc[tid] = acc;
    red[tid] = acc;
    __syncthreads();
    for (int off = 128; off > 0; off >>= 1) {
        if (tid < off) red[tid] = fmaxf(red[tid], red[tid + off]);
        __syncthreads();
    }
    float mx = red[0];
    __syncthreads();
    float e = __expf(sc[tid] - mx);
    red[tid] = e;
    __syncthreads();
    for (int off = 128; off > 0; off >>= 1) {
        if (tid < off) red[tid] += red[tid + off];
        __syncthreads();
    }
    float p = e / red[0];
    sc[tid] = p;
    prob[nb * Sdim + tid] = p;
    __syncthreads();
    for (int h = tid; h < Hdim; h += 256) {
        float c = 0.0f;
        for (int s2 = 0; s2 < Sdim; ++s2)
            c += sc[s2] * enc_out[((size_t)b * Sdim + s2) * Hdim + h];
        context[nb * Hdim + h] = c;
    }
}

// switch = sigmoid([h2|context|dec_in] . W_ratio + b)
__global__ __launch_bounds__(128) void switch_kernel(
    const float* __restrict__ hidden, const float* __restrict__ context,
    const float* __restrict__ dec_in, const float* __restrict__ Wr,
    const float* __restrict__ br, float* __restrict__ sw)
{
    __shared__ float red[128];
    const int nb = blockIdx.x, tid = threadIdx.x;
    float acc = 0.0f;
    for (int j = tid; j < Hdim; j += 128)
        acc += hidden[nb * Hdim + j] * Wr[j]
             + context[nb * Hdim + j] * Wr[Hdim + j]
             + dec_in[nb * Hdim + j] * Wr[2 * Hdim + j];
    red[tid] = acc;
    __syncthreads();
    for (int off = 64; off > 0; off >>= 1) {
        if (tid < off) red[tid] += red[tid + off];
        __syncthreads();
    }
    if (tid == 0) sw[nb] = sigm(red[0] + br[0]);
}

// all_gates = context(step0) @ W_gate^T + b
__global__ __launch_bounds__(128) void gateout_kernel(
    const float* __restrict__ context, const float* __restrict__ Wg,
    const float* __restrict__ bg, float* __restrict__ out_g)
{
    __shared__ float red[128];
    const int nb = blockIdx.x, tid = threadIdx.x;
    for (int g = 0; g < Gdim; ++g) {
        float acc = 0.0f;
        for (int j = tid; j < Hdim; j += 128) acc += context[nb * Hdim + j] * Wg[g * Hdim + j];
        red[tid] = acc;
        __syncthreads();
        for (int off = 64; off > 0; off >>= 1) {
            if (tid < off) red[tid] += red[tid + off];
            __syncthreads();
        }
        if (tid == 0) out_g[nb * Gdim + g] = red[0] + bg[g];
        __syncthreads();
    }
}

// ---------------------------------------------------------------------------
// Final: softmax over logits -> switch*p_vocab into d_out, scatter-add the
// pointer distribution, argmax (first-max), feed next dec_in = emb[pred].
// ---------------------------------------------------------------------------
__global__ __launch_bounds__(256) void final_kernel(
    const float* __restrict__ logits, const float* __restrict__ prob,
    const float* __restrict__ sw_arr, const int* __restrict__ story,
    const float* __restrict__ emb, float* __restrict__ out,
    float* __restrict__ dec_in, _Float16* __restrict__ dec_in_h,
    int* __restrict__ pred, int t)
{
    __shared__ float red[256];
    __shared__ int   redi[256];
    const int nb = blockIdx.x;
    const int b  = nb % Bdim;
    const int tid = threadIdx.x;
    const float* lrow = logits + (size_t)nb * Vdim;
    float* orow = out + ((size_t)nb * Tdim + t) * Vdim;

    float mx = -1e30f;
    for (int v = tid; v < Vdim; v += 256) mx = fmaxf(mx, lrow[v]);
    red[tid] = mx;
    __syncthreads();
    for (int off = 128; off > 0; off >>= 1) {
        if (tid < off) red[tid] = fmaxf(red[tid], red[tid + off]);
        __syncthreads();
    }
    mx = red[0];
    __syncthreads();
    float sum = 0.0f;
    for (int v = tid; v < Vdim; v += 256) sum += __expf(lrow[v] - mx);
    red[tid] = sum;
    __syncthreads();
    for (int off = 128; off > 0; off >>= 1) {
        if (tid < off) red[tid] += red[tid + off];
        __syncthreads();
    }
    const float inv = 1.0f / red[0];
    const float sw = sw_arr[nb];
    for (int v = tid; v < Vdim; v += 256)
        orow[v] = sw * __expf(lrow[v] - mx) * inv;
    __threadfence();
    __syncthreads();
    // pointer scatter: one lane per source position (S == 256 == blockDim)
    {
        float pv = (1.0f - sw) * prob[nb * Sdim + tid];
        atomicAdd(&orow[story[b * Sdim + tid]], pv);
    }
    __threadfence();
    __syncthreads();
    // argmax (first maximum)
    float bv = -1e30f; int bi = 0;
    for (int v = tid; v < Vdim; v += 256) {
        float x = orow[v];
        if (x > bv) { bv = x; bi = v; }
    }
    red[tid] = bv; redi[tid] = bi;
    __syncthreads();
    for (int off = 128; off > 0; off >>= 1) {
        if (tid < off) {
            float ov = red[tid + off]; int oi = redi[tid + off];
            if (ov > red[tid] || (ov == red[tid] && oi < redi[tid])) {
                red[tid] = ov; redi[tid] = oi;
            }
        }
        __syncthreads();
    }
    const int p = redi[0];
    for (int j = tid; j < Hdim; j += 256) {
        float e = emb[(size_t)p * Hdim + j];
        dec_in[nb * Hdim + j] = e;
        dec_in_h[nb * Kp + j] = (_Float16)e;
    }
    if (tid == 0) pred[nb] = p;
}

// ---------------------------------------------------------------------------
extern "C" void kernel_launch(void* const* d_in, const int* in_sizes, int n_in,
                              void* d_out, int out_size, void* d_ws, size_t ws_size,
                              hipStream_t stream)
{
    const int*   input_seqs = (const int*)d_in[0];
    const int*   story      = (const int*)d_in[1];
    const int*   dom_idx    = (const int*)d_in[2];
    const int*   slot_idx   = (const int*)d_in[3];
    const float* emb        = (const float*)d_in[4];
    const float* enc_Wi_f   = (const float*)d_in[5];
    const float* enc_Wh_f   = (const float*)d_in[6];
    const float* enc_bi_f   = (const float*)d_in[7];
    const float* enc_bh_f   = (const float*)d_in[8];
    const float* enc_Wi_b   = (const float*)d_in[9];
    const float* enc_Wh_b   = (const float*)d_in[10];
    const float* enc_bi_b   = (const float*)d_in[11];
    const float* enc_bh_b   = (const float*)d_in[12];
    const float* dec_Wi     = (const float*)d_in[13];
    const float* dec_Wh     = (const float*)d_in[14];
    const float* dec_bi     = (const float*)d_in[15];
    const float* dec_bh     = (const float*)d_in[16];
    const float* W_ratio_w  = (const float*)d_in[17];
    const float* W_ratio_b  = (const float*)d_in[18];
    const float* W_gate_w   = (const float*)d_in[19];
    const float* W_gate_b   = (const float*)d_in[20];
    const float* slot_table = (const float*)d_in[21];
    float* out = (float*)d_out;

    // ---- workspace carve-up (256B aligned) ----
    char* ws = (char*)d_ws;
    size_t off = 0;
    auto alloc = [&](size_t bytes) -> char* {
        char* p = ws + off;
        off += (bytes + 255) & ~(size_t)255;
        return p;
    };
    _Float16* x_h      = (_Float16*)alloc((size_t)SB * Kp * 2);
    _Float16* wif_h    = (_Float16*)alloc((size_t)H3 * Kp * 2);
    _Float16* wib_h    = (_Float16*)alloc((size_t)H3 * Kp * 2);
    _Float16* dwi_h    = (_Float16*)alloc((size_t)H3 * Kp * 2);
    _Float16* dwh_h    = (_Float16*)alloc((size_t)H3 * Kp * 2);
    _Float16* emb_h    = (_Float16*)alloc((size_t)Vdim * Kp * 2);
    float* gi_f        = (float*)alloc((size_t)SB * H3 * 4);
    float* gi_b        = (float*)alloc((size_t)SB * H3 * 4);
    float* enc_out     = (float*)alloc((size_t)Bdim * Sdim * Hdim * 4);
    float* hf          = (float*)alloc((size_t)Bdim * Hdim * 4);
    float* hb          = (float*)alloc((size_t)Bdim * Hdim * 4);
    float* hidden      = (float*)alloc((size_t)NB * Hdim * 4);
    _Float16* hidden_h = (_Float16*)alloc((size_t)NB * Kp * 2);
    float* dec_in      = (float*)alloc((size_t)NB * Hdim * 4);
    _Float16* dec_in_h = (_Float16*)alloc((size_t)NB * Kp * 2);
    float* gi_d        = (float*)alloc((size_t)NB * H3 * 4);
    float* gh_d        = (float*)alloc((size_t)NB * H3 * 4);
    float* logits      = (float*)alloc((size_t)NB * Vdim * 4);
    float* prob        = (float*)alloc((size_t)NB * Sdim * 4);
    float* context     = (float*)alloc((size_t)NB * Hdim * 4);
    float* sw          = (float*)alloc((size_t)NB * 4);
    int*   pred        = (int*)alloc((size_t)NB * 4);

    const int CB = 256;
    // ---- staging: f16 conversions (zero padded K=416) ----
    embed_x_kernel<<<(SB * Kp + CB - 1) / CB, CB, 0, stream>>>(input_seqs, emb, x_h);
    cvt_pad_kernel<<<(H3 * Kp + CB - 1) / CB, CB, 0, stream>>>(enc_Wi_f, wif_h, H3);
    cvt_pad_kernel<<<(H3 * Kp + CB - 1) / CB, CB, 0, stream>>>(enc_Wi_b, wib_h, H3);
    cvt_pad_kernel<<<(H3 * Kp + CB - 1) / CB, CB, 0, stream>>>(dec_Wi, dwi_h, H3);
    cvt_pad_kernel<<<(H3 * Kp + CB - 1) / CB, CB, 0, stream>>>(dec_Wh, dwh_h, H3);
    cvt_pad_kernel<<<((size_t)Vdim * Kp + CB - 1) / CB, CB, 0, stream>>>(emb, emb_h, Vdim);

    // ---- encoder: gi = x @ Wi^T + bi via WMMA (MT=4: 2048/64 = 32 strips) ----
    wmma_gemm_mt<4><<<dim3(H3 / 16, SB / 64), 32, 0, stream>>>(x_h, wif_h, enc_bi_f, gi_f, SB, H3, Kp);
    wmma_gemm_mt<4><<<dim3(H3 / 16, SB / 64), 32, 0, stream>>>(x_h, wib_h, enc_bi_b, gi_b, SB, H3, Kp);
    enc_scan_kernel<<<2, 1024, 0, stream>>>(gi_f, gi_b, enc_Wh_f, enc_bh_f, enc_Wh_b, enc_bh_b,
                                            enc_out, hf, hb);

    // ---- decoder init ----
    dec_init_kernel<<<(NB * Kp + CB - 1) / CB, CB, 0, stream>>>(
        hf, hb, slot_table, dom_idx, slot_idx, hidden, hidden_h, dec_in, dec_in_h);

    // ---- decoder steps (MT=5: 240/80 = 3 M-strips, B fragment reused 5x) ----
    for (int t = 0; t < Tdim; ++t) {
        wmma_gemm_mt<5><<<dim3(H3 / 16, NB / 80), 32, 0, stream>>>(dec_in_h, dwi_h, dec_bi, gi_d, NB, H3, Kp);
        wmma_gemm_mt<5><<<dim3(H3 / 16, NB / 80), 32, 0, stream>>>(hidden_h, dwh_h, dec_bh, gh_d, NB, H3, Kp);
        dec_gate_kernel<<<(NB * Hdim + CB - 1) / CB, CB, 0, stream>>>(gi_d, gh_d, hidden, hidden_h);
        // dominant GEMM: logits = h2 @ emb^T  (240 x 16000 x 400), emb strip reused 5x
        wmma_gemm_mt<5><<<dim3(Vdim / 16, NB / 80), 32, 0, stream>>>(hidden_h, emb_h, nullptr, logits, NB, Vdim, Kp);
        attn_kernel<<<NB, 256, 0, stream>>>(hidden, enc_out, prob, context);
        switch_kernel<<<NB, 128, 0, stream>>>(hidden, context, dec_in, W_ratio_w, W_ratio_b, sw);
        if (t == 0)
            gateout_kernel<<<NB, 128, 0, stream>>>(context, W_gate_w, W_gate_b,
                                                   out + (size_t)NB * Tdim * Vdim);
        final_kernel<<<NB, 256, 0, stream>>>(logits, prob, sw, story, emb, out,
                                             dec_in, dec_in_h, pred, t);
    }
}
